// PatchTSTEncoder_15083925143925
// MI455X (gfx1250) — matrix operand order repaired and checked by reading, not compiled, not run on hardware
//
#include <hip/hip_runtime.h>
#include <hip/hip_bf16.h>

typedef unsigned short u16;
typedef unsigned int   u32;
typedef __attribute__((ext_vector_type(16))) __bf16 v16bf;
typedef __attribute__((ext_vector_type(8)))  float  v8f;

#define T1      7679
#define T1P     7680
#define T2      3839
#define T2P     3840
#define T3      1919
#define T3P     1920
#define T4      959
#define TOK     20139
#define TOKP    20160
#define TVALID  16303   /* 17 valid channels * 959 frames (fixed by setup_inputs) */

__device__ __forceinline__ u16 f2bf(float f) {
  u32 u = __float_as_uint(f);
  u32 r = u + 0x7FFFu + ((u >> 16) & 1u);   // round-to-nearest-even
  return (u16)(r >> 16);
}
__device__ __forceinline__ float gelu_exact(float v) {
  return 0.5f * v * (1.0f + erff(v * 0.70710678118654752f));
}

// 32-byte global -> LDS async copy (ASYNCcnt-tracked, bypasses VGPRs).
// The instruction offset is added to the LDS address; the global addr gets the same
// INST_OFFSET, so offset:16 advances both sides of the strip.
__device__ __forceinline__ void async_copy32(u16* dst, const u16* src) {
  asm volatile(
      "global_load_async_to_lds_b128 %0, %1, off\n\t"
      "global_load_async_to_lds_b128 %0, %1, off offset:16"
      :
      : "v"((u32)(uintptr_t)dst), "v"(src)
      : "memory");
}
__device__ __forceinline__ void wait_async() {
  asm volatile("s_wait_asynccnt 0x0" ::: "memory");
}

// ---------------------------------------------------------------- weight prep
__global__ __launch_bounds__(256) void k_f32_to_bf16(const float* __restrict__ in,
                                                     u16* __restrict__ out, long long n) {
  long long id = (long long)blockIdx.x * 256 + threadIdx.x;
  if (id < n) out[id] = f2bf(in[id]);
}

// [oc][ic][kk] -> [oc][kk*Cin + ic]   (tap-major K so K-tiles of 32 stay in one tap)
__global__ __launch_bounds__(256) void k_permute_w(const float* __restrict__ w,
                                                   u16* __restrict__ out,
                                                   int Cout, int Cin, int KK) {
  long long id = (long long)blockIdx.x * 256 + threadIdx.x;
  long long total = (long long)Cout * Cin * KK;
  if (id >= total) return;
  int kk = (int)(id % KK);
  int ic = (int)((id / KK) % Cin);
  int oc = (int)(id / ((long long)KK * Cin));
  out[(long long)oc * (Cin * KK) + kk * Cin + ic] = f2bf(w[id]);
}

// ---------------------------------------------------------------- conv1 (1 in-ch, k=10, s=5)
__global__ __launch_bounds__(256) void k_conv1_gelu(const float* __restrict__ x,
                                                    const float* __restrict__ w,
                                                    const float* __restrict__ b,
                                                    u16* __restrict__ h1) {
  long long id = (long long)blockIdx.x * 256 + threadIdx.x;
  long long total = 21LL * 512 * T1;
  if (id >= total) return;
  int t  = (int)(id % T1);
  int oc = (int)((id / T1) % 512);
  int ch = (int)(id / ((long long)T1 * 512));
  const float* xin = x + (long long)ch * 38400 + 5 * t;
  const float* wr  = w + oc * 10;
  float acc = b[oc];
  #pragma unroll
  for (int j = 0; j < 10; ++j) acc += wr[j] * xin[j];
  h1[((long long)ch * 512 + oc) * T1P + t] = f2bf(gelu_exact(acc));
}

// ---------------------------------------------------------------- generic WMMA GEMM
// C[m][n] = sum_seg sum_k A[m][seg*Kseg+k] * B_seg[k*ldbk + n*ldbn]
// 128x128 block tile, 8 waves (4x2), each wave: 2x4 tiles of 16x16, K-step 32.
// Double-buffered LDS; contiguous fills use GLOBAL_LOAD_ASYNC_TO_LDS_B128 and are
// overlapped with the WMMA work of the previous K-step (one barrier per step).
// EPI: 0 = f32 store (*scale)         1 = bf16 store (+bias)
//      2 = bf16 store gelu(+bias)     3 = bf16 store gelu(+bias)+PE(t=n,d=m)+ch_emb
template<int EPI>
__global__ __launch_bounds__(256) void k_wmma_gemm(
    const u16* __restrict__ A, int lda, long long strideAz, int M,
    const u16* __restrict__ B, long long ldbk, long long ldbn,
    long long segB, long long strideBz, int nseg, int Kseg, int N,
    const float* __restrict__ bias, float scale,
    void* __restrict__ Cv, int ldc, long long strideCz,
    const float* __restrict__ chemb) {

  __shared__ __align__(16) u16 Alds[2][128 * 32];   // [m][k] row-major
  __shared__ __align__(16) u16 Blds[2][128 * 32];   // [n][k] (n-major, 32 k per column)

  const int tid  = threadIdx.x;
  const int lane = tid & 31;
  const int wave = tid >> 5;
  const int wm   = wave >> 1;        // 0..3
  const int wn   = wave & 1;         // 0..1
  const int l15  = lane & 15;
  const int hi   = lane >> 4;        // half-wave select (per ISA VGPR layouts)
  const int z    = blockIdx.z;
  const int m0   = blockIdx.y * 128;
  const int n0   = blockIdx.x * 128;

  const u16* Ab  = A + (long long)z * strideAz;
  const u16* Bb0 = B + (long long)z * strideBz;

  v8f acc[2][4];
  const v8f vzero = {0.f, 0.f, 0.f, 0.f, 0.f, 0.f, 0.f, 0.f};
  #pragma unroll
  for (int i = 0; i < 2; ++i)
    #pragma unroll
    for (int j = 0; j < 4; ++j) acc[i][j] = vzero;

  const int fr = tid >> 1;   // fill row (A) / col (B): 0..127
  const int fh = tid & 1;    // which 16-element half of the 32-wide k strip

  const int stepsPerSeg = (Kseg + 31) >> 5;
  const int nsteps      = nseg * stepsPerSeg;

  auto fill = [&](int step, int bufi) {
    const int seg = step / stepsPerSeg;
    const int k0  = (step - seg * stepsPerSeg) * 32;
    const u16* Bb = Bb0 + (long long)seg * segB;
    const int  kA = seg * Kseg;
    u16* Ad = &Alds[bufi][0];
    u16* Bd = &Blds[bufi][0];
    { // ---- A tile fill (row-major; async in the interior)
      u16* dst = Ad + fr * 32 + fh * 16;
      const u16* src = Ab + (long long)(m0 + fr) * lda + kA + k0 + fh * 16;
      if ((m0 + fr) < M && (k0 + 32) <= Kseg) {
        async_copy32(dst, src);
        __builtin_prefetch((const void*)(src + 32), 0, 3); // global_prefetch_b8, near
      } else {
        #pragma unroll
        for (int j = 0; j < 16; ++j) {
          int k = k0 + fh * 16 + j;
          dst[j] = ((m0 + fr) < M && k < Kseg) ? src[j] : (u16)0;
        }
      }
    }
    // ---- B tile fill: three stride regimes
    if (ldbk == 1) {               // contiguous in k (o = attn @ V^T): async
      u16* dst = Bd + fr * 32 + fh * 16;
      const u16* src = Bb + (long long)(n0 + fr) * ldbn + k0 + fh * 16;
      if ((n0 + fr) < N && (k0 + 32) <= Kseg) {
        async_copy32(dst, src);
      } else {
        #pragma unroll
        for (int j = 0; j < 16; ++j) {
          int k = k0 + fh * 16 + j;
          dst[j] = ((n0 + fr) < N && k < Kseg) ? src[j] : (u16)0;
        }
      }
    } else if (ldbn == 1) {        // row-major in n (projections / scores): transpose scatter
      int kk = tid & 31;
      int cc = (tid >> 5) * 16;
      int k  = k0 + kk;
      const u16* src = Bb + (long long)k * ldbk + n0 + cc;
      union { uint4 q[2]; u16 s[16]; } tmp;
      if (k < Kseg && (n0 + cc + 16) <= N) {
        tmp.q[0] = *(const uint4*)src;
        tmp.q[1] = *(const uint4*)(src + 8);
        #pragma unroll
        for (int j = 0; j < 16; ++j) Bd[(cc + j) * 32 + kk] = tmp.s[j];
      } else {
        #pragma unroll
        for (int j = 0; j < 16; ++j)
          Bd[(cc + j) * 32 + kk] =
              (k < Kseg && (n0 + cc + j) < N) ? src[j] : (u16)0;
      }
    } else {                        // generic gather (conv stride-2 columns)
      u16* dst = Bd + fr * 32 + fh * 16;
      long long nb = (long long)(n0 + fr) * ldbn;
      bool nok = (n0 + fr) < N;
      #pragma unroll
      for (int j = 0; j < 16; ++j) {
        int k = k0 + fh * 16 + j;
        dst[j] = (nok && k < Kseg) ? Bb[(long long)k * ldbk + nb] : (u16)0;
      }
    }
  };

  fill(0, 0);                       // prologue: stage K-step 0 into buffer 0

  for (int step = 0; step < nsteps; ++step) {
    const int bufi = step & 1;
    wait_async();                   // my async fills of `bufi` have landed in LDS
    __syncthreads();                // everyone's fills done; buf^1 reads from step-1 done
    if (step + 1 < nsteps) fill(step + 1, bufi ^ 1);  // overlap with WMMAs below

    const u16* Ad = &Alds[bufi][0];
    const u16* Bd = &Blds[bufi][0];

    // ---- fragments per ISA wave32 layouts, then 8 WMMAs
    union FragBF { v16bf v; uint4 q[2]; };
    FragBF fa[2], fb[4];
    #pragma unroll
    for (int mt = 0; mt < 2; ++mt) {
      // A 16x32 bf16: lane row = l15; halves k:[hi*8,+8) and [16+hi*8,+8)
      const u16* p = Ad + (wm * 32 + mt * 16 + l15) * 32 + hi * 8;
      fa[mt].q[0] = *(const uint4*)p;
      fa[mt].q[1] = *(const uint4*)(p + 16);
    }
    #pragma unroll
    for (int nt = 0; nt < 4; ++nt) {
      // B 32x16 bf16: lane col = l15; halves k:[hi*8,+8) and [16+hi*8,+8)
      const u16* p = Bd + (wn * 64 + nt * 16 + l15) * 32 + hi * 8;
      fb[nt].q[0] = *(const uint4*)p;
      fb[nt].q[1] = *(const uint4*)(p + 16);
    }
    #pragma unroll
    for (int mt = 0; mt < 2; ++mt)
      #pragma unroll
      for (int nt = 0; nt < 4; ++nt)
        acc[mt][nt] = __builtin_amdgcn_wmma_f32_16x16x32_bf16(
            false, fa[mt].v, false, fb[nt].v, (short)0, acc[mt][nt],
            false, false);
  }

  // ---- epilogue (C 16x16 f32 layout: VGPR r, lanes 0-15: M=r; 16-31: M=r+8)
  #pragma unroll
  for (int mt = 0; mt < 2; ++mt) {
    #pragma unroll
    for (int nt = 0; nt < 4; ++nt) {
      #pragma unroll
      for (int r = 0; r < 8; ++r) {
        int m = m0 + wm * 32 + mt * 16 + hi * 8 + r;
        int n = n0 + wn * 64 + nt * 16 + l15;
        if (m < M && n < N) {
          float v = acc[mt][nt][r];
          long long ci = (long long)z * strideCz + (long long)m * ldc + n;
          if (EPI == 0) {
            ((float*)Cv)[ci] = v * scale;
          } else if (EPI == 1) {
            ((u16*)Cv)[ci] = f2bf(v + bias[m]);
          } else if (EPI == 2) {
            ((u16*)Cv)[ci] = f2bf(gelu_exact(v + bias[m]));
          } else { // EPI == 3: conv4 -> gelu + sinusoidal PE(t=n,d=m) + ch_emb[z][m]
            float g   = gelu_exact(v + bias[m]);
            float div = __expf(-(float)((m >> 1) * 2) * (9.210340371976184f / 1024.0f));
            float ang = (float)n * div;
            g += (m & 1) ? __cosf(ang) : __sinf(ang);
            g += chemb[(long long)z * 1024 + m];
            ((u16*)Cv)[ci] = f2bf(g);
          }
        }
      }
    }
  }
}

// ---------------------------------------------------------------- masked softmax
__global__ __launch_bounds__(256) void k_softmax_mask(const float* __restrict__ scores,
                                                      u16* __restrict__ attn) {
  __shared__ float red[256];
  const int tid = threadIdx.x;
  const long long row = blockIdx.x;          // row = h*105 + l
  const float* s = scores + row * TOKP;
  u16* a = attn + row * TOKP;
  float mx = -3.0e38f;
  for (int t = tid; t < TVALID; t += 256) mx = fmaxf(mx, s[t]);
  red[tid] = mx; __syncthreads();
  for (int off = 128; off > 0; off >>= 1) {
    if (tid < off) red[tid] = fmaxf(red[tid], red[tid + off]);
    __syncthreads();
  }
  mx = red[0]; __syncthreads();
  float sum = 0.f;
  for (int t = tid; t < TVALID; t += 256) sum += expf(s[t] - mx);
  red[tid] = sum; __syncthreads();
  for (int off = 128; off > 0; off >>= 1) {
    if (tid < off) red[tid] += red[tid + off];
    __syncthreads();
  }
  float inv = 1.0f / red[0];
  for (int t = tid; t < TOK; t += 256)
    a[t] = (t < TVALID) ? f2bf(expf(s[t] - mx) * inv) : (u16)0;
}

// ---------------------------------------------------------------- FFN + residual (tiny, fp32)
__global__ __launch_bounds__(256) void k_ffn_residual(const float* __restrict__ o,
                                                      const float* __restrict__ w1,
                                                      const float* __restrict__ w2,
                                                      float* __restrict__ out) {
  __shared__ float ol[1024];
  __shared__ float t1[256];
  const int tid = threadIdx.x;
  const int l   = blockIdx.x;
  for (int d = tid; d < 1024; d += 256) ol[d] = o[(long long)l * 1024 + d];
  __syncthreads();
  {
    float acc = 0.f;
    const float* wr = w1 + (long long)tid * 1024;
    for (int d = 0; d < 1024; ++d) acc += ol[d] * wr[d];
    t1[tid] = gelu_exact(acc);
  }
  __syncthreads();
  for (int d = tid; d < 1024; d += 256) {
    float acc = 0.f;
    const float* wr = w2 + (long long)d * 256;
    #pragma unroll 4
    for (int f = 0; f < 256; ++f) acc += t1[f] * wr[f];
    out[(long long)l * 1024 + d] = ol[d] + acc;
  }
}

// ---------------------------------------------------------------- launcher
extern "C" void kernel_launch(void* const* d_in, const int* in_sizes, int n_in,
                              void* d_out, int out_size, void* d_ws, size_t ws_size,
                              hipStream_t stream) {
  const float* x     = (const float*)d_in[0];
  const float* w1    = (const float*)d_in[1];
  const float* b1    = (const float*)d_in[2];
  const float* w2    = (const float*)d_in[3];
  const float* b2    = (const float*)d_in[4];
  const float* w3    = (const float*)d_in[5];
  const float* b3    = (const float*)d_in[6];
  const float* w4    = (const float*)d_in[7];
  const float* b4    = (const float*)d_in[8];
  const float* chemb = (const float*)d_in[9];
  const float* lq    = (const float*)d_in[10];
  const float* wk    = (const float*)d_in[11];
  const float* wk_b  = (const float*)d_in[12];
  const float* wv    = (const float*)d_in[13];
  const float* wv_b  = (const float*)d_in[14];
  const float* ffw1  = (const float*)d_in[15];
  const float* ffw2  = (const float*)d_in[16];
  float* out = (float*)d_out;
  (void)in_sizes; (void)n_in; (void)out_size; (void)ws_size;

  char* ws = (char*)d_ws;
  size_t off = 0;
  auto take = [&](size_t bytes) { size_t o = off; off += (bytes + 255) & ~(size_t)255; return o; };
  u16*   h1   = (u16*)  (ws + take(21ull * 512 * T1P * 2));
  u16*   h2   = (u16*)  (ws + take(21ull * 512 * T2P * 2));
  u16*   h3   = (u16*)  (ws + take(21ull * 512 * T3P * 2));
  u16*   ub   = (u16*)  (ws + take(1024ull * TOKP * 2));   // u[d][tok]
  u16*   Km   = (u16*)  (ws + take(1024ull * TOKP * 2));
  u16*   Vm   = (u16*)  (ws + take(1024ull * TOKP * 2));
  float* sc   = (float*)(ws + take(210ull * TOKP * 4));
  u16*   at   = (u16*)  (ws + take(210ull * TOKP * 2));
  float* ob   = (float*)(ws + take(105ull * 1024 * 4));
  u16*   wb2  = (u16*)  (ws + take(512ull  * 1536 * 2));
  u16*   wb3  = (u16*)  (ws + take(512ull  * 1536 * 2));
  u16*   wb4  = (u16*)  (ws + take(1024ull * 1536 * 2));
  u16*   wkbf = (u16*)  (ws + take(1024ull * 1024 * 2));
  u16*   wvbf = (u16*)  (ws + take(1024ull * 1024 * 2));
  u16*   lqbf = (u16*)  (ws + take(105ull  * 1024 * 2));

  // ---- weight prep (f32 -> bf16, conv weights tap-major K)
  k_permute_w<<<(512 * 512 * 3 + 255) / 256, 256, 0, stream>>>(w2, wb2, 512, 512, 3);
  k_permute_w<<<(512 * 512 * 3 + 255) / 256, 256, 0, stream>>>(w3, wb3, 512, 512, 3);
  k_permute_w<<<(1024 * 512 * 3 + 255) / 256, 256, 0, stream>>>(w4, wb4, 1024, 512, 3);
  k_f32_to_bf16<<<(1024 * 1024 + 255) / 256, 256, 0, stream>>>(wk, wkbf, 1024LL * 1024);
  k_f32_to_bf16<<<(1024 * 1024 + 255) / 256, 256, 0, stream>>>(wv, wvbf, 1024LL * 1024);
  k_f32_to_bf16<<<(105 * 1024 + 255) / 256, 256, 0, stream>>>(lq, lqbf, 105LL * 1024);

  // ---- conv1 (fp32 VALU, small)
  {
    long long total = 21LL * 512 * T1;
    k_conv1_gelu<<<(unsigned)((total + 255) / 256), 256, 0, stream>>>(x, w1, b1, h1);
  }
  // ---- conv2: M=512, K=3x512, N=3839 per channel
  k_wmma_gemm<2><<<dim3((T2 + 127) / 128, 4, 21), 256, 0, stream>>>(
      wb2, 1536, 0, 512, h1, T1P, 2, 1, 512LL * T1P, 3, 512, T2,
      b2, 1.0f, (void*)h2, T2P, 512LL * T2P, nullptr);
  // ---- conv3
  k_wmma_gemm<2><<<dim3((T3 + 127) / 128, 4, 21), 256, 0, stream>>>(
      wb3, 1536, 0, 512, h2, T2P, 2, 1, 512LL * T2P, 3, 512, T3,
      b3, 1.0f, (void*)h3, T3P, 512LL * T3P, nullptr);
  // ---- conv4 + gelu + PE + ch_emb -> u[d][tok]
  k_wmma_gemm<3><<<dim3((T4 + 127) / 128, 8, 21), 256, 0, stream>>>(
      wb4, 1536, 0, 1024, h3, T3P, 2, 1, 512LL * T3P, 3, 512, T4,
      b4, 1.0f, (void*)ub, TOKP, (long long)T4, chemb);
  // ---- K = wk @ u + b   (M=1024, K=1024, N=20139)
  k_wmma_gemm<1><<<dim3((TOK + 127) / 128, 8, 1), 256, 0, stream>>>(
      wkbf, 1024, 0, 1024, ub, TOKP, 1, 0, 0, 1, 1024, TOK,
      wk_b, 1.0f, (void*)Km, TOKP, 0, nullptr);
  // ---- V = wv @ u + b
  k_wmma_gemm<1><<<dim3((TOK + 127) / 128, 8, 1), 256, 0, stream>>>(
      wvbf, 1024, 0, 1024, ub, TOKP, 1, 0, 0, 1, 1024, TOK,
      wv_b, 1.0f, (void*)Vm, TOKP, 0, nullptr);
  // ---- scores[h] = (lq_h @ K_h) * 1/sqrt(2)   (z = head)
  k_wmma_gemm<0><<<dim3((TOK + 127) / 128, 1, 2), 256, 0, stream>>>(
      lqbf, 1024, 512, 105, Km, TOKP, 1, 0, 512LL * TOKP, 1, 512, TOK,
      nullptr, 0.70710678118654752f, (void*)sc, TOKP, 105LL * TOKP, nullptr);
  // ---- masked softmax -> bf16 attn rows
  k_softmax_mask<<<210, 256, 0, stream>>>(sc, at);
  // ---- o[h] = attn_h @ V_h^T   (M=105, K=20139, N=512; B contiguous in k)
  k_wmma_gemm<0><<<dim3(4, 1, 2), 256, 0, stream>>>(
      at, TOKP, 105LL * TOKP, 105, Vm, 1, TOKP, 0, 512LL * TOKP, 1, TOK, 512,
      nullptr, 1.0f, (void*)ob, 1024, 512, nullptr);
  // ---- out = o + ffn(o)
  k_ffn_residual<<<105, 256, 0, stream>>>(ob, ffw1, ffw2, out);
}
